// QKVF_65532611002991
// MI455X (gfx1250) — matrix operand order repaired and checked by reading, not compiled, not run on hardware
//
#include <hip/hip_runtime.h>

typedef __attribute__((ext_vector_type(16))) _Float16 v16h;
typedef __attribute__((ext_vector_type(8)))  float    v8f;
typedef __attribute__((ext_vector_type(4)))  unsigned int v4u;
typedef __attribute__((ext_vector_type(8)))  int      v8i;
typedef __attribute__((ext_vector_type(4)))  int      v4i;

union HFrag { v16h v; uint4 q[2]; };

static const long TOKS = 65536;   // B*H*W = 16*64*64
#define LDSP 40                   // padded LDS row pitch (halfs): 80B = 20 banks

// ---------------- reduction helpers (wave32) ----------------
__device__ __forceinline__ float wave32_sum(float v) {
#pragma unroll
  for (int o = 16; o > 0; o >>= 1) v += __shfl_xor(v, o, 32);
  return v;
}
__device__ __forceinline__ float wave32_max(float v) {
#pragma unroll
  for (int o = 16; o > 0; o >>= 1) v = fmaxf(v, __shfl_xor(v, o, 32));
  return v;
}
__device__ __forceinline__ float block128_sum(float v, float* sh4) {
  v = wave32_sum(v);
  __syncthreads();
  if ((threadIdx.x & 31) == 0) sh4[threadIdx.x >> 5] = v;
  __syncthreads();
  return sh4[0] + sh4[1] + sh4[2] + sh4[3];
}
__device__ __forceinline__ float block256_sum(float v, float* sh8) {
  v = wave32_sum(v);
  __syncthreads();
  if ((threadIdx.x & 31) == 0) sh8[threadIdx.x >> 5] = v;
  __syncthreads();
  float r = 0.f;
#pragma unroll
  for (int i = 0; i < 8; ++i) r += sh8[i];
  return r;
}
__device__ __forceinline__ float block256_max(float v, float* sh8) {
  v = wave32_max(v);
  __syncthreads();
  if ((threadIdx.x & 31) == 0) sh8[threadIdx.x >> 5] = v;
  __syncthreads();
  float r = sh8[0];
#pragma unroll
  for (int i = 1; i < 8; ++i) r = fmaxf(r, sh8[i]);
  return r;
}

// ---------------- TDM: DMA one 128x32-f16 tile (row stride 128) into LDS ----------------
// LDS destination uses pitch LDSP (40 halfs = 80B): TDM pad = 4 DWORDs after every
// 16 DWORDs (one 32-half row).  pad_interval code 3 (=16 DW), pad_amount code 3 (=4 DW).
__device__ __forceinline__ void xat_tdm_load_a(const _Float16* gptr, unsigned lds_off) {
  unsigned long long ga = (unsigned long long)(uintptr_t)gptr;
  v4u g0 = { 1u,                                   // count=1 (valid descriptor)
             lds_off,                              // LDS byte address
             (unsigned)(ga & 0xFFFFFFFFull),       // global addr [31:0]
             (unsigned)((ga >> 32) & 0x01FFFFFFull) | (2u << 30) }; // [56:32] | type=2
  v8i g1 = { (int)((1u << 16) | (1u << 20) | (3u << 22) | (3u << 25)), // data_size=2B, pad on
             (int)(128u << 16),                    // tensor_dim0 = 128 (low16 into bits 63:48)
             0,                                    // tensor_dim0 hi / tensor_dim1 lo16 = 0
             (int)((32u << 16) | 1u),              // tensor_dim1 hi16 (65536>>16=1) | tile_dim0=32
             128,                                  // tile_dim1 = 128 rows
             128,                                  // tensor_dim0_stride = 128 elements
             0, 0 };
  v4i gz = { 0, 0, 0, 0 };
#if __clang_major__ >= 23
  v8i gz8 = { 0, 0, 0, 0, 0, 0, 0, 0 };
  __builtin_amdgcn_tensor_load_to_lds(g0, g1, gz, gz, gz8, 0);
#else
  __builtin_amdgcn_tensor_load_to_lds(g0, g1, gz, gz, 0);
#endif
}

// ---------------- layout converters ----------------
__global__ void xat_nchw_to_tok(const float* __restrict__ in, _Float16* __restrict__ out) {
  long o = (long)blockIdx.x * 256 + threadIdx.x;   // NCHW linear index
  int  c = (int)((o >> 12) & 127);
  long b = o >> 19;
  long p = o & 4095;
  out[((b << 12) + p) * 128 + c] = (_Float16)in[o];
}

__global__ void xat_add_f16(const _Float16* __restrict__ a, const _Float16* __restrict__ b,
                            _Float16* __restrict__ o) {
  long i = (long)blockIdx.x * 256 + threadIdx.x;
  o[i] = (_Float16)((float)a[i] + (float)b[i]);
}

// ---------------- conv3x3 pad1, implicit GEMM with WMMA ----------------
__global__ __launch_bounds__(256) void xat_conv3x3_wmma(
    const _Float16* __restrict__ in, const float* __restrict__ W,
    const float* __restrict__ bias, float* __restrict__ out, int Cin) {
  __shared__ _Float16 a_sh[128][LDSP];
  __shared__ _Float16 b_sh[128][LDSP];

  const int tid  = threadIdx.x;
  const int lane = tid & 31, wave = tid >> 5;
  const int wm = wave >> 1, wn = wave & 1;           // 4x2 wave grid
  const long t0 = (long)blockIdx.x * 128;

  const int r  = tid >> 1;        // token row in tile (also co for B staging)
  const int hf = tid & 1;
  const long t = t0 + r;
  const int p  = (int)(t & 4095);
  const int yy = p >> 6, xx = p & 63;

  v8f acc[2][4];
#pragma unroll
  for (int i = 0; i < 2; ++i)
#pragma unroll
    for (int j = 0; j < 4; ++j) acc[i][j] = {};

  for (int tap = 0; tap < 9; ++tap) {
    const int dy = tap / 3 - 1, dx = tap % 3 - 1;
    const int ty = yy + dy, tx = xx + dx;
    const bool valid = ((unsigned)ty < 64u) && ((unsigned)tx < 64u);
    const long tsrc = t + (long)dy * 64 + dx;

    for (int kc = 0; kc < Cin; kc += 32) {
      // ---- stage A tile (zero-padded halo)
      uint4 u0 = {}, u1 = {};
      if (valid) {
        const uint4* s = (const uint4*)(in + tsrc * Cin + kc + hf * 16);
        u0 = s[0]; u1 = s[1];
        if (kc + 32 < Cin)
          __builtin_prefetch((const void*)(in + tsrc * Cin + kc + 32), 0, 1);
      }
      *(uint4*)&a_sh[r][hf * 16]     = u0;
      *(uint4*)&a_sh[r][hf * 16 + 8] = u1;

      // ---- stage B tile transposed: b_sh[co][k] = W[co][kc+k][dy][dx]
      {
        _Float16 tmp[16];
#pragma unroll
        for (int j = 0; j < 16; ++j)
          tmp[j] = (_Float16)W[(((long)r * Cin + kc + hf * 16 + j) * 3 + (dy + 1)) * 3 + (dx + 1)];
        *(uint4*)&b_sh[r][hf * 16]     = *(uint4*)&tmp[0];
        *(uint4*)&b_sh[r][hf * 16 + 8] = *(uint4*)&tmp[8];
      }
      __syncthreads();

      const int hi = lane >> 4, ml = lane & 15;
      HFrag af[2], bf[4];
#pragma unroll
      for (int i = 0; i < 2; ++i) {
        const int row = wm * 32 + i * 16 + ml;
        af[i].q[0] = *(const uint4*)&a_sh[row][hi * 16];       // K 0..7 / 16..23
        af[i].q[1] = *(const uint4*)&a_sh[row][hi * 16 + 8];   // K 8..15 / 24..31
      }
#pragma unroll
      for (int j = 0; j < 4; ++j) {
        const int col = wn * 64 + j * 16 + ml;
        bf[j].q[0] = *(const uint4*)&b_sh[col][hi * 16];
        bf[j].q[1] = *(const uint4*)&b_sh[col][hi * 16 + 8];
      }
#pragma unroll
      for (int i = 0; i < 2; ++i)
#pragma unroll
        for (int j = 0; j < 4; ++j)
          acc[i][j] = __builtin_amdgcn_wmma_f32_16x16x32_f16(
              false, af[i].v, false, bf[j].v, (short)0, acc[i][j], false, false);
      __syncthreads();
    }
  }

  const int hi = lane >> 4, ml = lane & 15;
#pragma unroll
  for (int i = 0; i < 2; ++i)
#pragma unroll
    for (int j = 0; j < 4; ++j) {
      const int col = wn * 64 + j * 16 + ml;
      const float bv = bias[col];
#pragma unroll
      for (int e = 0; e < 8; ++e) {
        const int row = wm * 32 + i * 16 + hi * 8 + e;
        out[(t0 + row) * 128 + col] = acc[i][j][e] + bv;
      }
    }
}

// ---------------- generic [T,128]x[128,128] GEMM with WMMA (TDM-staged A) ----------------
// flags: 1 = write f32, 2 = write f16, 4 = add residual (f32)
__global__ __launch_bounds__(256) void xat_gemm_wmma(
    const _Float16* __restrict__ A, const float* __restrict__ W,
    const float* __restrict__ bias, float* __restrict__ outf,
    _Float16* __restrict__ outh, const float* __restrict__ resid, int flags) {
  __shared__ _Float16 a_sh[128][LDSP];
  __shared__ _Float16 b_sh[128][LDSP];

  const int tid  = threadIdx.x;
  const int lane = tid & 31, wave = tid >> 5;
  const int wm = wave >> 1, wn = wave & 1;
  const long t0 = (long)blockIdx.x * 128;
  const int r = tid >> 1, hf = tid & 1;
  const unsigned a_lds = (unsigned)(unsigned long long)(uintptr_t)(&a_sh[0][0]);

  v8f acc[2][4];
#pragma unroll
  for (int i = 0; i < 2; ++i)
#pragma unroll
    for (int j = 0; j < 4; ++j) acc[i][j] = {};

  for (int k0 = 0; k0 < 128; k0 += 32) {
    // ---- A tile via Tensor Data Mover (one wave issues the DMA)
    if (wave == 0)
      xat_tdm_load_a(A + t0 * 128 + k0, a_lds);

    // ---- B tile, transposed to [n][k] (W is row-major [k][n])
    {
      _Float16 tmp[16];
#pragma unroll
      for (int j = 0; j < 16; ++j)
        tmp[j] = (_Float16)W[(k0 + hf * 16 + j) * 128 + r];
      *(uint4*)&b_sh[r][hf * 16]     = *(uint4*)&tmp[0];
      *(uint4*)&b_sh[r][hf * 16 + 8] = *(uint4*)&tmp[8];
    }
    __builtin_amdgcn_s_wait_tensorcnt(0);
    __syncthreads();

    const int hi = lane >> 4, ml = lane & 15;
    HFrag af[2], bf[4];
#pragma unroll
    for (int i = 0; i < 2; ++i) {
      const int row = wm * 32 + i * 16 + ml;
      af[i].q[0] = *(const uint4*)&a_sh[row][hi * 16];
      af[i].q[1] = *(const uint4*)&a_sh[row][hi * 16 + 8];
    }
#pragma unroll
    for (int j = 0; j < 4; ++j) {
      const int col = wn * 64 + j * 16 + ml;
      bf[j].q[0] = *(const uint4*)&b_sh[col][hi * 16];
      bf[j].q[1] = *(const uint4*)&b_sh[col][hi * 16 + 8];
    }
#pragma unroll
    for (int i = 0; i < 2; ++i)
#pragma unroll
      for (int j = 0; j < 4; ++j)
        acc[i][j] = __builtin_amdgcn_wmma_f32_16x16x32_f16(
            false, af[i].v, false, bf[j].v, (short)0, acc[i][j], false, false);
    __syncthreads();
  }

  const int hi = lane >> 4, ml = lane & 15;
#pragma unroll
  for (int i = 0; i < 2; ++i)
#pragma unroll
    for (int j = 0; j < 4; ++j) {
      const int col = wn * 64 + j * 16 + ml;
      const float bv = bias[col];
#pragma unroll
      for (int e = 0; e < 8; ++e) {
        const int row = wm * 32 + i * 16 + hi * 8 + e;
        const long idx = (t0 + row) * 128 + col;
        float val = acc[i][j][e] + bv;
        if (flags & 4) val += resid[idx];
        if (flags & 1) outf[idx] = val;
        if (flags & 2) outh[idx] = (_Float16)val;
      }
    }
}

// ---------------- dual LayerNorm: z -> resar (in place) -> n (f16) ----------------
__global__ void xat_dual_ln(float* __restrict__ z, _Float16* __restrict__ n_out,
                            const float* __restrict__ g1, const float* __restrict__ b1,
                            const float* __restrict__ g2, const float* __restrict__ b2) {
  __shared__ float sh[4];
  const long t = blockIdx.x;
  const int c  = threadIdx.x;
  float v = z[t * 128 + c];
  float m = block128_sum(v, sh) * (1.f / 128.f);
  float d = v - m;
  float vr = block128_sum(d * d, sh) * (1.f / 128.f);
  float r1 = d * rsqrtf(vr + 1e-5f) * g1[c] + b1[c];
  z[t * 128 + c] = r1;
  float m2 = block128_sum(r1, sh) * (1.f / 128.f);
  float d2 = r1 - m2;
  float v2 = block128_sum(d2 * d2, sh) * (1.f / 128.f);
  n_out[t * 128 + c] = (_Float16)(d2 * rsqrtf(v2 + 1e-5f) * g2[c] + b2[c]);
}

// ---------------- softmax of k over spatial dim (per batch, per channel) ----------------
__global__ __launch_bounds__(256) void xat_softmax_k(float* __restrict__ k) {
  __shared__ float sh[8];
  const int bc = blockIdx.x;              // b*128 + c
  const long b = bc >> 7;
  const int  c = bc & 127;
  const long base = (b << 12) * 128 + c;
  float vals[16];
  float mx = -3.4e38f;
#pragma unroll
  for (int i = 0; i < 16; ++i) {
    vals[i] = k[base + (long)(i * 256 + threadIdx.x) * 128];
    mx = fmaxf(mx, vals[i]);
  }
  mx = block256_max(mx, sh);
  float s = 0.f;
#pragma unroll
  for (int i = 0; i < 16; ++i) { vals[i] = __expf(vals[i] - mx); s += vals[i]; }
  s = block256_sum(s, sh);
  const float inv = 1.f / s;
#pragma unroll
  for (int i = 0; i < 16; ++i)
    k[base + (long)(i * 256 + threadIdx.x) * 128] = vals[i] * inv;
}

// ---------------- softmax of q over per-head channel dim (32) ----------------
__global__ void xat_softmax_q(float* __restrict__ q) {
  const long t = blockIdx.x;
  const int c  = threadIdx.x;      // wave = one head
  float v = q[t * 128 + c];
  float mx = wave32_max(v);
  float e = __expf(v - mx);
  float s = wave32_sum(e);
  q[t * 128 + c] = e / s;
}

// ---------------- ctx[b,h] (32x32) = softmax(k)^T @ v over 4096 tokens ----------------
__global__ __launch_bounds__(256) void xat_ctx(const float* __restrict__ k,
                                               const _Float16* __restrict__ v,
                                               float* __restrict__ ctx) {
  __shared__ float ksh[128][33];
  __shared__ float vsh[128][33];
  const int bh = blockIdx.x;
  const long b = bh >> 2;
  const int  h = bh & 3;
  const int  i  = threadIdx.x >> 3;
  const int  jb = (threadIdx.x & 7) * 4;
  float acc[4] = {0.f, 0.f, 0.f, 0.f};
  for (int chunk = 0; chunk < 32; ++chunk) {
    const long base = (b << 12) + chunk * 128;
#pragma unroll
    for (int e = 0; e < 16; ++e) {
      const int idx = threadIdx.x * 16 + e;
      const int tt = idx >> 5, cc = idx & 31;
      ksh[tt][cc] = k[(base + tt) * 128 + h * 32 + cc];
      vsh[tt][cc] = (float)v[(base + tt) * 128 + h * 32 + cc];
    }
    __syncthreads();
    for (int tt = 0; tt < 128; ++tt) {
      const float a = ksh[tt][i];
#pragma unroll
      for (int jj = 0; jj < 4; ++jj) acc[jj] += a * vsh[tt][jb + jj];
    }
    __syncthreads();
  }
#pragma unroll
  for (int jj = 0; jj < 4; ++jj)
    ctx[(long)bh * 1024 + i * 32 + jb + jj] = acc[jj];
}

// ---------------- agg[t,c] = sum_kk ctx[h][kk][v] * q[t][h*32+kk] ----------------
__global__ __launch_bounds__(256) void xat_agg(const float* __restrict__ ctx,
                                               const float* __restrict__ q,
                                               _Float16* __restrict__ agg) {
  __shared__ float csh[4096];
  const long t0 = (long)blockIdx.x * 128;
  const long b  = t0 >> 12;
  for (int i = threadIdx.x; i < 4096; i += 256) csh[i] = ctx[b * 4096 + i];
  __syncthreads();
  const int tl = threadIdx.x >> 1;
  const int half = threadIdx.x & 1;
  const long t = t0 + tl;
#pragma unroll
  for (int hh = 0; hh < 2; ++hh) {
    const int h = half * 2 + hh;
    float qr[32];
#pragma unroll
    for (int kk = 0; kk < 32; ++kk) qr[kk] = q[t * 128 + h * 32 + kk];
    for (int vj = 0; vj < 32; ++vj) {
      float acc = 0.f;
#pragma unroll
      for (int kk = 0; kk < 32; ++kk) acc += csh[h * 1024 + kk * 32 + vj] * qr[kk];
      agg[t * 128 + h * 32 + vj] = (_Float16)acc;
    }
  }
}

// ---------------- final LN + gating + relu outputs + cat build ----------------
__global__ void xat_final_mix(const float* __restrict__ fina1, const float* __restrict__ fina2,
                              const float* __restrict__ x, const float* __restrict__ y,
                              const float* __restrict__ g, const float* __restrict__ bb,
                              float* __restrict__ out1, float* __restrict__ out2,
                              _Float16* __restrict__ cat) {
  __shared__ float sh[4];
  const long t = blockIdx.x;
  const int  c = threadIdx.x;
  const long b = t >> 12;
  const long p = t & 4095;
  const long nchw = ((b * 128 + c) << 12) + p;

  float a1 = fina1[t * 128 + c];
  float m  = block128_sum(a1, sh) * (1.f / 128.f);
  float d  = a1 - m;
  float vr = block128_sum(d * d, sh) * (1.f / 128.f);
  float ln1 = d * rsqrtf(vr + 1e-5f) * g[c] + bb[c];
  float xv = x[nchw];
  float f1 = ln1 * xv + xv;

  float a2 = fina2[t * 128 + c];
  float m2 = block128_sum(a2, sh) * (1.f / 128.f);
  float d2 = a2 - m2;
  float v2 = block128_sum(d2 * d2, sh) * (1.f / 128.f);
  float ln2 = d2 * rsqrtf(v2 + 1e-5f) * g[c] + bb[c];
  float yv = y[nchw];
  float f2 = ln2 * yv + yv;

  out1[nchw] = fmaxf(f1, 0.f);
  out2[nchw] = fmaxf(f2, 0.f);
  cat[t * 256 + c]       = (_Float16)(f1 * f2);
  cat[t * 256 + 128 + c] = (_Float16)(f1 + f2);
}

// ---------------- BN(eval) + relu, token-major f32 -> NCHW ----------------
__global__ void xat_bn_relu(const float* __restrict__ z, const float* __restrict__ g,
                            const float* __restrict__ bb, const float* __restrict__ mean,
                            const float* __restrict__ var, float* __restrict__ out) {
  long o = (long)blockIdx.x * 256 + threadIdx.x;   // NCHW linear index
  int  c = (int)((o >> 12) & 127);
  long b = o >> 19;
  long p = o & 4095;
  float zv = z[((b << 12) + p) * 128 + c];
  float r = (zv - mean[c]) * rsqrtf(var[c] + 1e-5f) * g[c] + bb[c];
  out[o] = fmaxf(r, 0.f);
}

// ============================ launch ============================
extern "C" void kernel_launch(void* const* d_in, const int* in_sizes, int n_in,
                              void* d_out, int out_size, void* d_ws, size_t ws_size,
                              hipStream_t stream) {
  const float* x       = (const float*)d_in[0];
  const float* y       = (const float*)d_in[1];
  const float* pah_w   = (const float*)d_in[2];
  const float* pah_b   = (const float*)d_in[3];
  const float* pah_g   = (const float*)d_in[4];
  const float* pah_bb  = (const float*)d_in[5];
  const float* norm_g  = (const float*)d_in[6];
  const float* norm_b  = (const float*)d_in[7];
  const float* keys_w  = (const float*)d_in[8];
  const float* keys_b  = (const float*)d_in[9];
  const float* qry_w   = (const float*)d_in[10];
  const float* qry_b   = (const float*)d_in[11];
  const float* val_w   = (const float*)d_in[12];
  const float* val_b   = (const float*)d_in[13];
  const float* rep_w   = (const float*)d_in[14];
  const float* rep_b   = (const float*)d_in[15];
  const float* cow     = (const float*)d_in[16];
  const float* cob     = (const float*)d_in[17];
  const float* bn_g    = (const float*)d_in[18];
  const float* bn_b    = (const float*)d_in[19];
  const float* bn_mean = (const float*)d_in[20];
  const float* bn_var  = (const float*)d_in[21];

  const size_t F32B = (size_t)TOKS * 128 * 4;   // 32 MB
  const size_t F16B = (size_t)TOKS * 128 * 2;   // 16 MB
  char* ws = (char*)d_ws;
  _Float16* xt  = (_Float16*)(ws);                         // later reused as agg1
  _Float16* yt  = (_Float16*)(ws + F16B);                  // later reused as agg2
  float*    z1  = (float*)(ws + 2 * F16B);                 // conv-out / resar1 / fina1
  float*    z2  = (float*)(ws + 2 * F16B + F32B);          // conv-out / resar2 / fina2
  _Float16* n1h = (_Float16*)(ws + 2 * F16B + 2 * F32B);
  _Float16* n2h = (_Float16*)(ws + 3 * F16B + 2 * F32B);
  _Float16* sh_ = (_Float16*)(ws + 4 * F16B + 2 * F32B);
  float*    kf  = (float*)(ws + 5 * F16B + 2 * F32B);
  float*    q1  = (float*)(ws + 5 * F16B + 3 * F32B);      // later reused as co_conv out
  float*    q2  = (float*)(ws + 5 * F16B + 4 * F32B);
  _Float16* v1h = (_Float16*)(ws + 5 * F16B + 5 * F32B);
  _Float16* v2h = (_Float16*)(ws + 6 * F16B + 5 * F32B);
  float*    ctx1 = (float*)(ws + 7 * F16B + 5 * F32B);
  float*    ctx2 = (float*)(ws + 7 * F16B + 5 * F32B + 262144);
  _Float16* cat  = (_Float16*)(ws + 7 * F16B + 5 * F32B + 524288);

  float* out1 = (float*)d_out;
  float* out2 = out1 + (long)TOKS * 128;
  float* out3 = out2 + (long)TOKS * 128;

  const int EW_BLKS = (int)(TOKS * 128 / 256);   // 32768
  const int GEMM_BLKS = (int)(TOKS / 128);       // 512

  // 1) NCHW -> token f16
  xat_nchw_to_tok<<<EW_BLKS, 256, 0, stream>>>(x, xt);
  xat_nchw_to_tok<<<EW_BLKS, 256, 0, stream>>>(y, yt);

  // 2) patch-embed conv3x3 (WMMA implicit GEMM)
  xat_conv3x3_wmma<<<GEMM_BLKS, 256, 0, stream>>>(xt, pah_w, pah_b, z1, 128);
  xat_conv3x3_wmma<<<GEMM_BLKS, 256, 0, stream>>>(yt, pah_w, pah_b, z2, 128);

  // 3) dual LN: z -> resar (in place), n (f16)
  xat_dual_ln<<<(int)TOKS, 128, 0, stream>>>(z1, n1h, pah_g, pah_bb, norm_g, norm_b);
  xat_dual_ln<<<(int)TOKS, 128, 0, stream>>>(z2, n2h, pah_g, pah_bb, norm_g, norm_b);

  // 4) shared key input s = n1 + n2
  xat_add_f16<<<EW_BLKS, 256, 0, stream>>>(n1h, n2h, sh_);

  // 5) QKV linears (WMMA, TDM-staged A)
  xat_gemm_wmma<<<GEMM_BLKS, 256, 0, stream>>>(sh_, keys_w, keys_b, kf, nullptr, nullptr, 1);
  xat_gemm_wmma<<<GEMM_BLKS, 256, 0, stream>>>(n2h, qry_w, qry_b, q1, nullptr, nullptr, 1);
  xat_gemm_wmma<<<GEMM_BLKS, 256, 0, stream>>>(n1h, qry_w, qry_b, q2, nullptr, nullptr, 1);
  xat_gemm_wmma<<<GEMM_BLKS, 256, 0, stream>>>(n1h, val_w, val_b, nullptr, v1h, nullptr, 2);
  xat_gemm_wmma<<<GEMM_BLKS, 256, 0, stream>>>(n2h, val_w, val_b, nullptr, v2h, nullptr, 2);

  // 6) softmaxes
  xat_softmax_k<<<16 * 128, 256, 0, stream>>>(kf);
  xat_softmax_q<<<(int)TOKS, 128, 0, stream>>>(q1);
  xat_softmax_q<<<(int)TOKS, 128, 0, stream>>>(q2);

  // 7) per-head 32x32 context
  xat_ctx<<<64, 256, 0, stream>>>(kf, v1h, ctx1);
  xat_ctx<<<64, 256, 0, stream>>>(kf, v2h, ctx2);

  // 8) agg = ctx^T @ q (token-parallel), reuse xt/yt as agg buffers
  xat_agg<<<GEMM_BLKS, 256, 0, stream>>>(ctx1, q1, xt);
  xat_agg<<<GEMM_BLKS, 256, 0, stream>>>(ctx2, q2, yt);

  // 9) reproj + residual -> fina (in place over resar)
  xat_gemm_wmma<<<GEMM_BLKS, 256, 0, stream>>>(xt, rep_w, rep_b, z1, nullptr, z1, 1 | 4);
  xat_gemm_wmma<<<GEMM_BLKS, 256, 0, stream>>>(yt, rep_w, rep_b, z2, nullptr, z2, 1 | 4);

  // 10) final LN + gating + relu outputs + cat
  xat_final_mix<<<(int)TOKS, 128, 0, stream>>>(z1, z2, x, y, norm_g, norm_b, out1, out2, cat);

  // 11) co_conv 256->128 (WMMA implicit GEMM), out into q1 buffer
  xat_conv3x3_wmma<<<GEMM_BLKS, 256, 0, stream>>>(cat, cow, cob, q1, 256);

  // 12) BN(eval) + relu -> out3 (NCHW)
  xat_bn_relu<<<EW_BLKS, 256, 0, stream>>>(q1, bn_g, bn_b, bn_mean, bn_var, out3);
}
